// AttentionClassifier_6811818131558
// MI455X (gfx1250) — matrix operand order repaired and checked by reading, not compiled, not run on hardware
//
#include <hip/hip_runtime.h>
#include <hip/hip_fp16.h>
#include <hip/hip_bf16.h>

#define DEV __device__ __forceinline__

typedef __attribute__((ext_vector_type(16))) _Float16 v16h;
typedef __attribute__((ext_vector_type(8)))  float    v8f;

// ---------------- problem dims ----------------
static const int B_   = 64;
static const int T_   = 800;
static const int E_   = 100;
static const int EP_  = 128;   // E padded to multiple of 32 (WMMA K)
static const int H1_  = 100;
static const int G1_  = 300;   // 3*H1 gates
static const int G1P_ = 320;   // padded to 20 N-tiles
static const int A1_  = 200;   // h1 feature dim (fwd ++ bwd)
static const int A1P_ = 224;   // padded K for attention scores
static const int DH_  = 400;   // GRU2 input dim (h1 ++ x_hat)
static const int DHP_ = 416;   // padded K
static const int H2_  = 400;
static const int G2_  = 1200;  // 3*H2
static const int G2P_ = 1216;  // 76 N-tiles
static const int O2_  = 800;   // h2 feature dim (fwd ++ bwd), already K%32==0

// ---------------- WMMA fragment loaders (wave32, 16x16x32 f16) ----------------
// A fragment: 16x32, rows striped per-lane. lane l: row=l&15, group g=l>>4.
// Per lane this is one contiguous 16-half (32B) run: element e covers k = 16*g + e.
DEV v16h wmma_load_a(const _Float16* src, long ld) {
  const int l = threadIdx.x & 31;
  const int row = l & 15, g = l >> 4;
  const _Float16* p = src + (long)row * ld + 16 * g;
  v16h a;
#pragma unroll
  for (int e = 0; e < 16; ++e) {
    const int v = e >> 1;
    const int k = e;
    a[e] = p[k];
  }
  return a;
}
// B fragment from k-major storage src[k][n] (ld = row stride in elements):
// lane l: col=l&15, g=l>>4; element e covers k = 16*g + e.
DEV v16h wmma_load_b(const _Float16* src, long ld) {
  const int l = threadIdx.x & 31;
  const int col = l & 15, g = l >> 4;
  const _Float16* p = src + (long)(16 * g) * ld + col;
  v16h b;
#pragma unroll
  for (int e = 0; e < 16; ++e) b[e] = p[(long)e * ld];
  return b;
}
// B fragment = transpose of a row-major [n][k] matrix (e.g. h1 for scores).
DEV v16h wmma_load_bt(const _Float16* src, long ld) {
  const int l = threadIdx.x & 31;
  const int col = l & 15, g = l >> 4;
  const _Float16* p = src + (long)col * ld + 16 * g;
  v16h b;
#pragma unroll
  for (int e = 0; e < 16; ++e) b[e] = p[e];
  return b;
}
// Same but from f32 source with column (n) guard (for Wf: 50 valid rows).
DEV v16h wmma_load_bt_f32_guard(const float* src, long ld, int valid_cols) {
  const int l = threadIdx.x & 31;
  const int col = l & 15, g = l >> 4;
  const float* p = src + (long)col * ld + 16 * g;
  v16h b;
#pragma unroll
  for (int e = 0; e < 16; ++e)
    b[e] = (col < valid_cols) ? (_Float16)p[e] : (_Float16)0.f;
  return b;
}
DEV v8f wmma32(v16h a, v16h b, v8f c) {
  return __builtin_amdgcn_wmma_f32_16x16x32_f16(false, a, false, b, (short)0, c,
                                                false, false);
}
DEV float sigmoidf_(float x) { return 1.f / (1.f + __expf(-x)); }

// ---------------- weight prep: W[N,K] f32 -> D[Kpad][Npad] f16 (W^T, padded) --
__global__ void prep_weight_kernel(const float* __restrict__ W,
                                   _Float16* __restrict__ D,
                                   int N, int K, int Npad, int Kpad) {
  int idx = blockIdx.x * blockDim.x + threadIdx.x;
  if (idx >= Npad * Kpad) return;
  int n = idx % Npad, k = idx / Npad;
  D[idx] = (n < N && k < K) ? (_Float16)W[(long)n * K + k] : (_Float16)0.f;
}

// ---------------- embedding gather (emb[1] is already zero = pad idx) --------
__global__ void embed_kernel(const float* __restrict__ emb,
                             const int* __restrict__ x,
                             _Float16* __restrict__ xeh) {
  long idx = (long)blockIdx.x * blockDim.x + threadIdx.x;
  if (idx >= (long)B_ * T_ * EP_) return;
  int e = (int)(idx & (EP_ - 1));
  long bt = idx >> 7;  // EP_ = 128
  xeh[idx] = (e < E_) ? (_Float16)emb[(long)x[bt] * E_ + e] : (_Float16)0.f;
}

// zero the K-pad columns of h1buf (cols 200..223) so attention K-pad is clean
__global__ void h1pad_kernel(_Float16* __restrict__ h1buf) {
  const int PADC = A1P_ - A1_;
  long idx = (long)blockIdx.x * blockDim.x + threadIdx.x;
  if (idx >= (long)B_ * T_ * PADC) return;
  long bt = idx / PADC;
  int c = (int)(idx % PADC);
  h1buf[bt * A1P_ + A1_ + c] = (_Float16)0.f;
}

// ---------------- GRU layer 1 (persistent recurrence, WMMA per step) ---------
// grid.x = 8 : dir = bx>>2, batch chunk of 16 = (bx&3)*16. 320 thr = 10 waves.
// Each wave owns 2 N-tiles; A fragments loaded once per k-chunk and reused.
__global__ void __launch_bounds__(320) gru1_kernel(
    const _Float16* __restrict__ xeh,
    const _Float16* __restrict__ wihT_f, const _Float16* __restrict__ whhT_f,
    const float* __restrict__ bih_f, const float* __restrict__ bhh_f,
    const _Float16* __restrict__ wihT_b, const _Float16* __restrict__ whhT_b,
    const float* __restrict__ bih_b, const float* __restrict__ bhh_b,
    _Float16* __restrict__ h1buf) {
  const int dir = blockIdx.x >> 2;
  const int b0 = (blockIdx.x & 3) * 16;
  const _Float16* wihT = dir ? wihT_b : wihT_f;
  const _Float16* whhT = dir ? whhT_b : whhT_f;
  const float* bih = dir ? bih_b : bih_f;
  const float* bhh = dir ? bhh_b : bhh_f;

  __shared__ _Float16 hstate_h[16][EP_];   // f16 copy for WMMA A (zero k-pad)
  __shared__ float    hstate_f[16][H1_];   // f32 master state
  __shared__ float    xg[16][G1P_];
  __shared__ float    hg[16][G1P_];

  for (int i = threadIdx.x; i < 16 * EP_; i += 320)
    (&hstate_h[0][0])[i] = (_Float16)0.f;
  for (int i = threadIdx.x; i < 16 * H1_; i += 320)
    (&hstate_f[0][0])[i] = 0.f;
  __syncthreads();

  const int w = threadIdx.x >> 5;
  const int lane = threadIdx.x & 31;
  const int g = lane >> 4;
  // hoist loop-invariant biases (one scalar pair per owned tile)
  const int nt0 = 2 * w, nt1 = 2 * w + 1;
  const int col0 = nt0 * 16 + (lane & 15);
  const int col1 = nt1 * 16 + (lane & 15);
  const float bx0 = (col0 < G1_) ? bih[col0] : 0.f;
  const float bh0 = (col0 < G1_) ? bhh[col0] : 0.f;
  const float bx1 = (col1 < G1_) ? bih[col1] : 0.f;
  const float bh1 = (col1 < G1_) ? bhh[col1] : 0.f;

  for (int s = 0; s < T_; ++s) {
    const int t = dir ? (T_ - 1 - s) : s;
    const _Float16* abase = xeh + ((long)b0 * T_ + t) * EP_;
    if (s + 1 < T_) {  // prefetch next step's activations (global_prefetch_b8)
      const int tn = dir ? (T_ - 2 - s) : (s + 1);
      __builtin_prefetch(xeh + ((long)(b0 + (lane & 15)) * T_ + tn) * EP_, 0, 3);
    }
    // GEMM: xg = x_t @ Wih^T + bih ; hg = h @ Whh^T + bhh  (M=16, N=320, K=128)
    v8f ax0 = {}, ax1 = {}, ah0 = {}, ah1 = {};
#pragma unroll
    for (int kc = 0; kc < EP_ / 32; ++kc) {
      v16h fa_x = wmma_load_a(abase + kc * 32, (long)T_ * EP_);
      v16h fa_h = wmma_load_a(&hstate_h[0][0] + kc * 32, EP_);
      ax0 = wmma32(fa_x, wmma_load_b(wihT + (long)kc * 32 * G1P_ + nt0 * 16, G1P_), ax0);
      ax1 = wmma32(fa_x, wmma_load_b(wihT + (long)kc * 32 * G1P_ + nt1 * 16, G1P_), ax1);
      ah0 = wmma32(fa_h, wmma_load_b(whhT + (long)kc * 32 * G1P_ + nt0 * 16, G1P_), ah0);
      ah1 = wmma32(fa_h, wmma_load_b(whhT + (long)kc * 32 * G1P_ + nt1 * 16, G1P_), ah1);
    }
#pragma unroll
    for (int i = 0; i < 8; ++i) {
      xg[8 * g + i][col0] = ax0[i] + bx0;
      hg[8 * g + i][col0] = ah0[i] + bh0;
      xg[8 * g + i][col1] = ax1[i] + bx1;
      hg[8 * g + i][col1] = ah1[i] + bh1;
    }
    __syncthreads();
    // gates (torch order r,z,n): h' = (1-z)*tanh(x_n + r*gh_n) + z*h
    for (int u = threadIdx.x; u < 16 * H1_; u += 320) {
      const int ri = u / H1_, j = u - ri * H1_;
      const float r = sigmoidf_(xg[ri][j] + hg[ri][j]);
      const float z = sigmoidf_(xg[ri][H1_ + j] + hg[ri][H1_ + j]);
      const float n = tanhf(xg[ri][2 * H1_ + j] + r * hg[ri][2 * H1_ + j]);
      const float hn = (1.f - z) * n + z * hstate_f[ri][j];
      hstate_f[ri][j] = hn;
      hstate_h[ri][j] = (_Float16)hn;
      h1buf[((long)(b0 + ri) * T_ + t) * A1P_ + dir * H1_ + j] = (_Float16)hn;
    }
    __syncthreads();
  }
}

// ---------------- fused self-attention: one block per (b, 16-query tile) ----
__global__ void __launch_bounds__(512) attn_kernel(
    const _Float16* __restrict__ h1buf, const int* __restrict__ mask,
    _Float16* __restrict__ hbuf) {
  const int b = blockIdx.y;
  const int t0 = blockIdx.x * 16;
  const int w = threadIdx.x >> 5;
  const int lane = threadIdx.x & 31;
  __shared__ float    S[16][T_];      // raw scores -> masked exps
  __shared__ _Float16 Sh[16][T_];     // normalized weights (WMMA A operand)
  __shared__ float    xh[16][A1P_];   // x_hat tile

  const _Float16* h1b = h1buf + (long)b * T_ * A1P_;
  // phase 1: S = h1_tile @ h1^T (K=224). Query-tile A fragments are loop-
  // invariant across column tiles: keep all 7 in registers.
  {
    v16h afrag[A1P_ / 32];
#pragma unroll
    for (int kc = 0; kc < A1P_ / 32; ++kc)
      afrag[kc] = wmma_load_a(h1b + (long)t0 * A1P_ + kc * 32, A1P_);
    for (int ct = w; ct < T_ / 16; ct += 16) {
      v8f acc = {};
#pragma unroll
      for (int kc = 0; kc < A1P_ / 32; ++kc) {
        v16h bb = wmma_load_bt(h1b + (long)(ct * 16) * A1P_ + kc * 32, A1P_);
        acc = wmma32(afrag[kc], bb, acc);
      }
      const int col = ct * 16 + (lane & 15);
      const int g = lane >> 4;
#pragma unroll
      for (int i = 0; i < 8; ++i) {
        float v = acc[i];
        if (t0 + 8 * g + i == col) v = 0.f;  // off-diagonal mask (pre-softmax)
        S[8 * g + i][col] = v;
      }
    }
  }
  __syncthreads();
  // phase 2: per-row softmax + diag/pad masking + eps-renorm (wave w = row w)
  {
    const int r = w;
    const float mr = (float)mask[b * T_ + t0 + r];
    float mx = -1e30f;
    for (int c = lane; c < T_; c += 32) mx = fmaxf(mx, S[r][c]);
    for (int off = 16; off; off >>= 1) mx = fmaxf(mx, __shfl_xor(mx, off, 32));
    float sum = 0.f;
    for (int c = lane; c < T_; c += 32) {
      float pv = __expf(S[r][c] - mx);
      if (t0 + r == c) pv = 0.f;                 // off-diagonal (post-softmax)
      pv *= mr * (float)mask[b * T_ + c];        // pad mask
      S[r][c] = pv;
      sum += pv;
    }
    for (int off = 16; off; off >>= 1) sum += __shfl_xor(sum, off, 32);
    const float inv = 1.f / (sum + 1e-13f);
    for (int c = lane; c < T_; c += 32) Sh[r][c] = (_Float16)(S[r][c] * inv);
  }
  __syncthreads();
  // phase 3: x_hat = Sh @ h1   (M=16, N=224, K=800 = 25 chunks)
  if (w < A1P_ / 16) {
    v8f acc = {};
    for (int kc = 0; kc < T_ / 32; ++kc) {
      v16h a  = wmma_load_a(&Sh[0][0] + kc * 32, T_);
      v16h bb = wmma_load_b(h1b + (long)(kc * 32) * A1P_ + w * 16, A1P_);
      acc = wmma32(a, bb, acc);
    }
    const int col = w * 16 + (lane & 15);
    const int g = lane >> 4;
#pragma unroll
    for (int i = 0; i < 8; ++i) xh[8 * g + i][col] = acc[i];
  }
  __syncthreads();
  // phase 4: h = [h1 ++ x_hat ++ 0pad] into GRU2 input buffer
  for (int idx = threadIdx.x; idx < 16 * DHP_; idx += 512) {
    const int r = idx / DHP_, c = idx - r * DHP_;
    _Float16 v;
    if (c < A1_)      v = h1b[(long)(t0 + r) * A1P_ + c];
    else if (c < DH_) v = (_Float16)xh[r][c - A1_];
    else              v = (_Float16)0.f;
    hbuf[((long)b * T_ + t0 + r) * DHP_ + c] = v;
  }
}

// ---------------- GRU layer 2 (M=16, N=1216, K=416 per step) -----------------
// 16 waves; wave w owns tiles {w, w+16, ...} (5 for w<12, else 4; wave-uniform
// so EXEC stays all-ones around WMMA). A fragments shared across tiles.
__global__ void __launch_bounds__(512) gru2_kernel(
    const _Float16* __restrict__ hbuf,
    const _Float16* __restrict__ wihT_f, const _Float16* __restrict__ whhT_f,
    const float* __restrict__ bih_f, const float* __restrict__ bhh_f,
    const _Float16* __restrict__ wihT_b, const _Float16* __restrict__ whhT_b,
    const float* __restrict__ bih_b, const float* __restrict__ bhh_b,
    _Float16* __restrict__ h2buf) {
  const int dir = blockIdx.x >> 2;
  const int b0 = (blockIdx.x & 3) * 16;
  const _Float16* wihT = dir ? wihT_b : wihT_f;
  const _Float16* whhT = dir ? whhT_b : whhT_f;
  const float* bih = dir ? bih_b : bih_f;
  const float* bhh = dir ? bhh_b : bhh_f;

  __shared__ _Float16 hstate_h[16][DHP_];  // 13.3 KB
  __shared__ float    hstate_f[16][H2_];   // 25.6 KB
  __shared__ _Float16 xg[16][G2P_];        // 38.9 KB (input gates tolerate f16)
  __shared__ float    hg[16][G2P_];        // 77.8 KB (needed at f32 for r*gh_n)

  for (int i = threadIdx.x; i < 16 * DHP_; i += 512)
    (&hstate_h[0][0])[i] = (_Float16)0.f;
  for (int i = threadIdx.x; i < 16 * H2_; i += 512)
    (&hstate_f[0][0])[i] = 0.f;
  __syncthreads();

  const int w = threadIdx.x >> 5;
  const int lane = threadIdx.x & 31;
  const int g = lane >> 4;
  const int NT = G2P_ / 16;               // 76
  const int ntile_cnt = (w < NT % 16) ? (NT / 16 + 1) : (NT / 16);  // 5 or 4
  // hoisted per-tile column indices and biases
  int cols[5];
  float bx_c[5], bh_c[5];
#pragma unroll
  for (int ti = 0; ti < 5; ++ti) {
    const int nt = w + 16 * ti;
    const int col = nt * 16 + (lane & 15);
    cols[ti] = col;
    const bool ok = (ti < ntile_cnt) && (col < G2_);
    bx_c[ti] = ok ? bih[col] : 0.f;
    bh_c[ti] = ok ? bhh[col] : 0.f;
  }

  for (int s = 0; s < T_; ++s) {
    const int t = dir ? (T_ - 1 - s) : s;
    const _Float16* abase = hbuf + ((long)b0 * T_ + t) * DHP_;
    if (s + 1 < T_) {
      const int tn = dir ? (T_ - 2 - s) : (s + 1);
      __builtin_prefetch(hbuf + ((long)(b0 + (lane & 15)) * T_ + tn) * DHP_, 0, 3);
    }
    v8f accx[5], acch[5];
#pragma unroll
    for (int ti = 0; ti < 5; ++ti) { accx[ti] = {}; acch[ti] = {}; }
    for (int kc = 0; kc < DHP_ / 32; ++kc) {
      v16h fa_x = wmma_load_a(abase + kc * 32, (long)T_ * DHP_);
      v16h fa_h = wmma_load_a(&hstate_h[0][0] + kc * 32, DHP_);
#pragma unroll
      for (int ti = 0; ti < 5; ++ti) {
        if (ti < ntile_cnt) {  // wave-uniform guard
          const int nt = w + 16 * ti;
          accx[ti] = wmma32(fa_x,
              wmma_load_b(wihT + (long)kc * 32 * G2P_ + nt * 16, G2P_), accx[ti]);
          acch[ti] = wmma32(fa_h,
              wmma_load_b(whhT + (long)kc * 32 * G2P_ + nt * 16, G2P_), acch[ti]);
        }
      }
    }
#pragma unroll
    for (int ti = 0; ti < 5; ++ti) {
      if (ti < ntile_cnt) {
        const int col = cols[ti];
#pragma unroll
        for (int i = 0; i < 8; ++i) {
          xg[8 * g + i][col] = (_Float16)(accx[ti][i] + bx_c[ti]);
          hg[8 * g + i][col] = acch[ti][i] + bh_c[ti];
        }
      }
    }
    __syncthreads();
    for (int u = threadIdx.x; u < 16 * H2_; u += 512) {
      const int ri = u / H2_, j = u - ri * H2_;
      const float r = sigmoidf_((float)xg[ri][j] + hg[ri][j]);
      const float z = sigmoidf_((float)xg[ri][H2_ + j] + hg[ri][H2_ + j]);
      const float n = tanhf((float)xg[ri][2 * H2_ + j] + r * hg[ri][2 * H2_ + j]);
      const float hn = (1.f - z) * n + z * hstate_f[ri][j];
      hstate_f[ri][j] = hn;
      hstate_h[ri][j] = (_Float16)hn;
      h2buf[((long)(b0 + ri) * T_ + t) * O2_ + dir * H2_ + j] = (_Float16)hn;
    }
    __syncthreads();
  }
}

// ---------------- final-attention pooling: one block per batch row -----------
__global__ void __launch_bounds__(256) pool_kernel(
    const _Float16* __restrict__ h2buf, const float* __restrict__ cv,
    const int* __restrict__ mask, _Float16* __restrict__ pooled) {
  const int b = blockIdx.x;
  __shared__ float u[T_];
  __shared__ float red[8];
  __shared__ float sval;
  const _Float16* h2b = h2buf + (long)b * T_ * O2_;
  for (int t = threadIdx.x; t < T_; t += 256) {
    float acc = 0.f;
    const _Float16* row = h2b + (long)t * O2_;
    for (int e = 0; e < O2_; ++e) acc += (float)row[e] * cv[e];
    u[t] = acc;
  }
  __syncthreads();
  float mx = -1e30f;
  for (int t = threadIdx.x; t < T_; t += 256) mx = fmaxf(mx, u[t]);
  for (int off = 16; off; off >>= 1) mx = fmaxf(mx, __shfl_xor(mx, off, 32));
  if ((threadIdx.x & 31) == 0) red[threadIdx.x >> 5] = mx;
  __syncthreads();
  if (threadIdx.x == 0) {
    float m = red[0];
    for (int i = 1; i < 8; ++i) m = fmaxf(m, red[i]);
    sval = m;
  }
  __syncthreads();
  mx = sval;
  __syncthreads();
  float sum = 0.f;
  for (int t = threadIdx.x; t < T_; t += 256) {
    float a = __expf(u[t] - mx) * (float)mask[b * T_ + t];
    u[t] = a;
    sum += a;
  }
  for (int off = 16; off; off >>= 1) sum += __shfl_xor(sum, off, 32);
  if ((threadIdx.x & 31) == 0) red[threadIdx.x >> 5] = sum;
  __syncthreads();
  if (threadIdx.x == 0) {
    float ssum = 0.f;
    for (int i = 0; i < 8; ++i) ssum += red[i];
    sval = ssum;
  }
  __syncthreads();
  const float inv = 1.f / (sval + 1e-13f);
  for (int e = threadIdx.x; e < O2_; e += 256) {
    float acc = 0.f;
    for (int t = 0; t < T_; ++t) acc += u[t] * (float)h2b[(long)t * O2_ + e];
    pooled[b * O2_ + e] = (_Float16)(acc * inv);
  }
}

// ---------------- final linear: [64,800] @ [800,50] + bias via WMMA ----------
__global__ void __launch_bounds__(512) final_kernel(
    const _Float16* __restrict__ pooled, const float* __restrict__ Wf,
    const float* __restrict__ bf, float* __restrict__ out) {
  const int w = threadIdx.x >> 5;
  const int lane = threadIdx.x & 31;
  const int mt = w >> 2, nt = w & 3;  // 4x4 tile grid = 16 waves
  v8f acc = {};
  for (int kc = 0; kc < O2_ / 32; ++kc) {
    v16h a  = wmma_load_a(pooled + (long)(mt * 16) * O2_ + kc * 32, O2_);
    v16h bb = wmma_load_bt_f32_guard(Wf + (long)(nt * 16) * O2_ + kc * 32, O2_,
                                     50 - nt * 16);
    acc = wmma32(a, bb, acc);
  }
  const int col = nt * 16 + (lane & 15);
  const int g = lane >> 4;
  if (col < 50) {
#pragma unroll
    for (int i = 0; i < 8; ++i)
      out[(mt * 16 + 8 * g + i) * 50 + col] = acc[i] + bf[col];
  }
}

// ---------------- host-side orchestration ------------------------------------
extern "C" void kernel_launch(void* const* d_in, const int* in_sizes, int n_in,
                              void* d_out, int out_size, void* d_ws,
                              size_t ws_size, hipStream_t stream) {
  (void)in_sizes; (void)n_in; (void)out_size; (void)ws_size;
  const float* emb   = (const float*)d_in[0];
  const float* Wih1f = (const float*)d_in[1];
  const float* Whh1f = (const float*)d_in[2];
  const float* bih1f = (const float*)d_in[3];
  const float* bhh1f = (const float*)d_in[4];
  const float* Wih1b = (const float*)d_in[5];
  const float* Whh1b = (const float*)d_in[6];
  const float* bih1b = (const float*)d_in[7];
  const float* bhh1b = (const float*)d_in[8];
  const float* Wih2f = (const float*)d_in[9];
  const float* Whh2f = (const float*)d_in[10];
  const float* bih2f = (const float*)d_in[11];
  const float* bhh2f = (const float*)d_in[12];
  const float* Wih2b = (const float*)d_in[13];
  const float* Whh2b = (const float*)d_in[14];
  const float* bih2b = (const float*)d_in[15];
  const float* bhh2b = (const float*)d_in[16];
  const float* cv    = (const float*)d_in[17];
  const float* Wf    = (const float*)d_in[18];
  const float* bf    = (const float*)d_in[19];
  const int*   x     = (const int*)d_in[20];
  /* d_in[21] = x_len : unused by reference */
  const int*   mask  = (const int*)d_in[22];
  float* out = (float*)d_out;

  char* p = (char*)d_ws;
  auto alloc = [&](size_t bytes) -> char* {
    char* r = p;
    p += (bytes + 255) & ~(size_t)255;
    return r;
  };
  _Float16* xeh    = (_Float16*)alloc((size_t)B_ * T_ * EP_ * 2);
  _Float16* h1buf  = (_Float16*)alloc((size_t)B_ * T_ * A1P_ * 2);
  _Float16* hbuf   = (_Float16*)alloc((size_t)B_ * T_ * DHP_ * 2);
  _Float16* h2buf  = (_Float16*)alloc((size_t)B_ * T_ * O2_ * 2);
  _Float16* pooled = (_Float16*)alloc((size_t)B_ * O2_ * 2);
  _Float16* w_ih1f = (_Float16*)alloc((size_t)EP_ * G1P_ * 2);
  _Float16* w_hh1f = (_Float16*)alloc((size_t)EP_ * G1P_ * 2);
  _Float16* w_ih1b = (_Float16*)alloc((size_t)EP_ * G1P_ * 2);
  _Float16* w_hh1b = (_Float16*)alloc((size_t)EP_ * G1P_ * 2);
  _Float16* w_ih2f = (_Float16*)alloc((size_t)DHP_ * G2P_ * 2);
  _Float16* w_hh2f = (_Float16*)alloc((size_t)DHP_ * G2P_ * 2);
  _Float16* w_ih2b = (_Float16*)alloc((size_t)DHP_ * G2P_ * 2);
  _Float16* w_hh2b = (_Float16*)alloc((size_t)DHP_ * G2P_ * 2);

  auto prep = [&](const float* W, _Float16* D, int N, int K, int Np, int Kp) {
    int total = Np * Kp;
    prep_weight_kernel<<<(total + 255) / 256, 256, 0, stream>>>(W, D, N, K, Np, Kp);
  };
  prep(Wih1f, w_ih1f, G1_, E_,  G1P_, EP_);
  prep(Whh1f, w_hh1f, G1_, H1_, G1P_, EP_);
  prep(Wih1b, w_ih1b, G1_, E_,  G1P_, EP_);
  prep(Whh1b, w_hh1b, G1_, H1_, G1P_, EP_);
  prep(Wih2f, w_ih2f, G2_, DH_, G2P_, DHP_);
  prep(Whh2f, w_hh2f, G2_, H2_, G2P_, DHP_);
  prep(Wih2b, w_ih2b, G2_, DH_, G2P_, DHP_);
  prep(Whh2b, w_hh2b, G2_, H2_, G2P_, DHP_);

  {
    long total = (long)B_ * T_ * EP_;
    embed_kernel<<<(int)((total + 255) / 256), 256, 0, stream>>>(emb, x, xeh);
  }
  {
    long total = (long)B_ * T_ * (A1P_ - A1_);
    h1pad_kernel<<<(int)((total + 255) / 256), 256, 0, stream>>>(h1buf);
  }
  gru1_kernel<<<8, 320, 0, stream>>>(xeh, w_ih1f, w_hh1f, bih1f, bhh1f,
                                     w_ih1b, w_hh1b, bih1b, bhh1b, h1buf);
  attn_kernel<<<dim3(T_ / 16, B_), 512, 0, stream>>>(h1buf, mask, hbuf);
  gru2_kernel<<<8, 512, 0, stream>>>(hbuf, w_ih2f, w_hh2f, bih2f, bhh2f,
                                     w_ih2b, w_hh2b, bih2b, bhh2b, h2buf);
  pool_kernel<<<B_, 256, 0, stream>>>(h2buf, cv, mask, pooled);
  final_kernel<<<1, 512, 0, stream>>>(pooled, Wf, bf, out);
}
